// MultiHeadAttention_43241730736830
// MI455X (gfx1250) — compile-verified, matrix operand order checked by reading, not run on hardware
//
#include <hip/hip_runtime.h>
#include <hip/hip_bf16.h>

#define D_MODEL 1024
#define NHEAD   16
#define D_K     64
#define SEQ     2048
#define BATCH   2

typedef __attribute__((ext_vector_type(16))) __bf16 v16bf;
typedef __attribute__((ext_vector_type(8)))  __bf16 v8bf;
typedef __attribute__((ext_vector_type(8)))  float  v8f;
typedef int v4i_gcc __attribute__((vector_size(16)));  // matches builtin param type

#if defined(__has_builtin)
#if __has_builtin(__builtin_amdgcn_global_load_async_to_lds_b128) && \
    __has_builtin(__builtin_amdgcn_s_wait_asynccnt)
#define USE_ASYNC_LDS 1
#endif
#if __has_builtin(__builtin_amdgcn_permlane16)
#define USE_PERMLANE16 1
#endif
#endif

// ---------------------------------------------------------------------------
// WMMA wrapper: D = A(16x32 bf16) * B(32x16 bf16) + C(16x16 f32)
// ---------------------------------------------------------------------------
__device__ __forceinline__ v8f wmma_bf16(v16bf a, v16bf b, v8f c) {
  return __builtin_amdgcn_wmma_f32_16x16x32_bf16(
      /*neg_a=*/false, a, /*neg_b=*/false, b,
      /*c_mod=*/(short)0, c, /*reuse_a=*/false, /*reuse_b=*/false);
}

__device__ __forceinline__ v8f zero_v8f() {
  v8f z = {0.f, 0.f, 0.f, 0.f, 0.f, 0.f, 0.f, 0.f};
  return z;
}

// ---------------------------------------------------------------------------
// Fragment loader for a row-major 16xK tile (A-layout; also B when the source
// stores B^T row-major — true for W, Kh, Vt and the LDS staging tiles).
// lane<16 : row (lane&15), K = {k0..k0+7, k0+16..k0+23}
// lane>=16: same row,      K = {k0+8..+15, k0+24..+31}
// ---------------------------------------------------------------------------
__device__ __forceinline__ v16bf load_frag(const __bf16* __restrict__ base,
                                           int ld, int row0, int k0) {
  const int lane = (int)(threadIdx.x & 31u);
  const __bf16* p =
      base + (size_t)(row0 + (lane & 15)) * (size_t)ld + k0 + ((lane < 16) ? 0 : 8);
  v8bf lo = *(const v8bf*)p;
  v8bf hi = *(const v8bf*)(p + 16);
  v16bf f;
#pragma unroll
  for (int i = 0; i < 8; ++i) { f[i] = lo[i]; f[i + 8] = hi[i]; }
  return f;
}

// ---------------------------------------------------------------------------
// max-reduce across the 16 lanes of each half-wave (rows of the C layout).
// Prefer VALU v_permlane16_b32 butterfly (no LDS round-trip) over ds_bpermute.
// ---------------------------------------------------------------------------
__device__ __forceinline__ float halfrow_max(float x) {
#ifdef USE_PERMLANE16
  // nibble selectors: lane i reads lane i^m within its 16-lane row
  const unsigned selLo[4] = {0x67452301u, 0x54761032u, 0x32107654u, 0xFEDCBA98u};
  const unsigned selHi[4] = {0xEFCDAB89u, 0xDCFE98BAu, 0xBA98FEDCu, 0x76543210u};
  unsigned v = __float_as_uint(x);
#pragma unroll
  for (int i = 0; i < 4; ++i) {
    unsigned y = __builtin_amdgcn_permlane16(v, v, selLo[i], selHi[i], false, false);
    v = __float_as_uint(fmaxf(__uint_as_float(v), __uint_as_float(y)));
  }
  return __uint_as_float(v);
#else
#pragma unroll
  for (int xm = 8; xm >= 1; xm >>= 1) x = fmaxf(x, __shfl_xor(x, xm, 32));
  return x;
#endif
}

#ifdef USE_ASYNC_LDS
// per-lane 16-byte async copy global -> LDS (tracked by ASYNCcnt)
__device__ __forceinline__ void async_cp16(const __bf16* g, __bf16* l) {
  __builtin_amdgcn_global_load_async_to_lds_b128(
      (__attribute__((address_space(1))) v4i_gcc*)(unsigned long long)g,
      (__attribute__((address_space(3))) v4i_gcc*)(unsigned)(unsigned long long)l,
      0, 0);
}
#endif

// ---------------------------------------------------------------------------
// fp32 -> bf16 conversion
// ---------------------------------------------------------------------------
__global__ void f32_to_bf16(const float* __restrict__ in,
                            __bf16* __restrict__ out, int n) {
  int i = blockIdx.x * blockDim.x + threadIdx.x;
  if (i < n) out[i] = (__bf16)in[i];
}

// ---------------------------------------------------------------------------
// Projection GEMM: out[m, n] = sum_k X[m,k] * W[n,k] + bias[n]
// MODE 0: X = (S,B,D) bf16, batch via blockIdx.z; out -> Qh/Kh (b,h,s,d) bf16
// MODE 1: same, but out -> Vt (b,h,d,s) bf16 (transposed store)
// MODE 2: X = (S*B, D) bf16 rows; out -> fp32 (S,B,D) flat
// Block = 256 threads (8 waves); each wave computes a 32x64 output tile.
// ---------------------------------------------------------------------------
template <int MODE>
__global__ __launch_bounds__(256) void proj_kernel(
    const __bf16* __restrict__ X, const __bf16* __restrict__ W,
    const float* __restrict__ bias, void* __restrict__ OutP) {
  const int wave = (int)(threadIdx.x >> 5);
  const int lane = (int)(threadIdx.x & 31u);
  const int row0 = (int)blockIdx.x * 256 + wave * 32;
  const int n0   = (int)blockIdx.y * 64;
  const int b    = (int)blockIdx.z;

  const __bf16* abase = (MODE == 2) ? X : (X + b * D_MODEL);
  const int lda       = (MODE == 2) ? D_MODEL : (BATCH * D_MODEL);

  v8f acc[2][4];
#pragma unroll
  for (int mm = 0; mm < 2; ++mm)
#pragma unroll
    for (int t = 0; t < 4; ++t) acc[mm][t] = zero_v8f();

  for (int k0 = 0; k0 < D_MODEL; k0 += 32) {
    const v16bf a0 = load_frag(abase, lda, row0, k0);
    const v16bf a1 = load_frag(abase, lda, row0 + 16, k0);
#pragma unroll
    for (int t = 0; t < 4; ++t) {
      const v16bf bf = load_frag(W, D_MODEL, n0 + t * 16, k0);
      acc[0][t] = wmma_bf16(a0, bf, acc[0][t]);
      acc[1][t] = wmma_bf16(a1, bf, acc[1][t]);
    }
  }

  const int col   = lane & 15;
  const int rbase = (lane >> 4) * 8;
#pragma unroll
  for (int mm = 0; mm < 2; ++mm) {
#pragma unroll
    for (int t = 0; t < 4; ++t) {
      const int n  = n0 + t * 16 + col;
      const float bv = bias[n];
      const int h  = n >> 6;
      const int dl = n & 63;
#pragma unroll
      for (int r = 0; r < 8; ++r) {
        const int m = row0 + mm * 16 + rbase + r;
        const float val = acc[mm][t][r] + bv;
        if (MODE == 0) {
          __bf16* out = (__bf16*)OutP;
          out[(size_t)((b * NHEAD + h) * SEQ + m) * D_K + dl] = (__bf16)val;
        } else if (MODE == 1) {
          __bf16* out = (__bf16*)OutP;
          out[(size_t)((b * NHEAD + h) * D_K + dl) * SEQ + m] = (__bf16)val;
        } else {
          float* out = (float*)OutP;
          out[(size_t)m * D_MODEL + n] = val;
        }
      }
    }
  }
}

// ---------------------------------------------------------------------------
// Flash attention per (b,h): block = 128 threads (4 waves); each wave owns
// 16 query rows, streams keys in chunks of 32 with online softmax (exp2
// domain). Row sums via WMMA against an all-ones B fragment. K/V tiles for
// the next chunk are staged into per-wave double-buffered LDS with
// GLOBAL_LOAD_ASYNC_TO_LDS_B128 (ASYNCcnt), overlapping with current WMMAs.
// ---------------------------------------------------------------------------
__global__ __launch_bounds__(128) void attn_kernel(
    const __bf16* __restrict__ Qh, const __bf16* __restrict__ Kh,
    const __bf16* __restrict__ Vt, const float* __restrict__ mask,
    __bf16* __restrict__ Ctx) {
  __shared__ __align__(16) __bf16 Pbuf[4 * 16 * 32];  // per-wave P staging
#ifdef USE_ASYNC_LDS
  __shared__ __align__(16) __bf16 Kst[4][2][32 * 64];  // per-wave K tiles
  __shared__ __align__(16) __bf16 Vst[4][2][64 * 32];  // per-wave V tiles
#endif

  const int wave = (int)(threadIdx.x >> 5);
  const int lane = (int)(threadIdx.x & 31u);
  const int bh   = (int)blockIdx.y;
  const int b    = bh >> 4;
  const int h    = bh & 15;
  const int s0   = (int)blockIdx.x * 64 + wave * 16;

  const __bf16* qb = Qh + (size_t)bh * SEQ * D_K;
  const __bf16* kb = Kh + (size_t)bh * SEQ * D_K;
  const __bf16* vb = Vt + (size_t)bh * D_K * SEQ;

  const v16bf qf0 = load_frag(qb, D_K, s0, 0);
  const v16bf qf1 = load_frag(qb, D_K, s0, 32);

  v16bf onesf;
#pragma unroll
  for (int i = 0; i < 16; ++i) onesf[i] = (__bf16)1.0f;

  const int col   = lane & 15;
  const int rbase = (lane >> 4) * 8;
  const float LOG2E  = 1.44269504088896340736f;
  const float scale2 = 0.125f * LOG2E;  // (1/sqrt(Dk)) * log2(e)

  float mrow[8];
  v8f accO[4] = {zero_v8f(), zero_v8f(), zero_v8f(), zero_v8f()};
  v8f accL    = zero_v8f();
#pragma unroll
  for (int r = 0; r < 8; ++r) mrow[r] = -__builtin_inff();

  __bf16* pb = &Pbuf[wave * 512];

#ifdef USE_ASYNC_LDS
  auto stage = [&](int buf, int j0) {
    // K tile: 32x64 bf16, contiguous 4KB from Kh
    {
      const __bf16* g = kb + (size_t)j0 * D_K;
      __bf16* l = &Kst[wave][buf][0];
#pragma unroll
      for (int i = 0; i < 8; ++i) {
        const int off = (i * 32 + lane) * 8;  // elements; 16B per lane
        async_cp16(g + off, l + off);
      }
    }
    // V tile: 64 rows x 32 cols from Vt (row stride SEQ)
    {
      const int rr = lane >> 2;          // 0..7
      const int cc = (lane & 3) * 8;     // 0,8,16,24
#pragma unroll
      for (int i = 0; i < 8; ++i) {
        const int r = i * 8 + rr;
        async_cp16(vb + (size_t)r * SEQ + j0 + cc, &Vst[wave][buf][r * 32 + cc]);
      }
    }
  };
  stage(0, 0);
#endif

  for (int j0 = 0; j0 < SEQ; j0 += 32) {
    const __bf16* kc_base;  int kc_ld, kc_row0;   // score B-frags (K rows)
    const __bf16* vc_base;  int vc_ld, vc_k0;     // P@V B-frags (Vt rows)
#ifdef USE_ASYNC_LDS
    const int buf = (j0 >> 5) & 1;
    if (j0 + 32 < SEQ) {
      stage(buf ^ 1, j0 + 32);
      __builtin_amdgcn_s_wait_asynccnt(16);  // current chunk's 16 done
    } else {
      __builtin_amdgcn_s_wait_asynccnt(0);
    }
    kc_base = &Kst[wave][buf][0]; kc_ld = 64;  kc_row0 = 0;
    vc_base = &Vst[wave][buf][0]; vc_ld = 32;  vc_k0 = 0;
#else
    {
      const int jn = (j0 + 32 < SEQ) ? (j0 + 32) : j0;
      __builtin_prefetch(kb + (size_t)(jn + (lane & 31)) * D_K, 0, 1);
      __builtin_prefetch(vb + (size_t)(lane & 31) * SEQ + jn, 0, 1);
    }
    kc_base = kb; kc_ld = D_K; kc_row0 = j0;
    vc_base = vb; vc_ld = SEQ; vc_k0 = j0;
#endif

    // ---- scores (exp2 domain): two 16x16 subtiles, 2 WMMA k-steps each ----
    v8f s[2];
#pragma unroll
    for (int sub = 0; sub < 2; ++sub) {
      v8f c = zero_v8f();
      c = wmma_bf16(qf0, load_frag(kc_base, kc_ld, kc_row0 + sub * 16, 0), c);
      c = wmma_bf16(qf1, load_frag(kc_base, kc_ld, kc_row0 + sub * 16, 32), c);
      const int j = j0 + sub * 16 + col;
#pragma unroll
      for (int r = 0; r < 8; ++r) {
        const int si = s0 + rbase + r;
        c[r] = c[r] * scale2 + mask[(size_t)si * SEQ + j] * LOG2E;
      }
      s[sub] = c;
    }

    // ---- online softmax: rowmax via VALU permlane butterfly ----
    float alpha[8];
#pragma unroll
    for (int r = 0; r < 8; ++r) {
      const float cm = halfrow_max(fmaxf(s[0][r], s[1][r]));
      const float mn = fmaxf(mrow[r], cm);
      alpha[r] = exp2f(mrow[r] - mn);
      mrow[r]  = mn;
    }
#pragma unroll
    for (int sub = 0; sub < 2; ++sub) {
#pragma unroll
      for (int r = 0; r < 8; ++r) {
        const float p = exp2f(s[sub][r] - mrow[r]);
        pb[(rbase + r) * 32 + sub * 16 + col] = (__bf16)p;  // ds_store_b16
      }
    }
#pragma unroll
    for (int r = 0; r < 8; ++r) {
      accL[r] *= alpha[r];
#pragma unroll
      for (int tt = 0; tt < 4; ++tt) accO[tt][r] *= alpha[r];
    }

    // ---- reload P in A-fragment order (same-wave DS ops are in-order) ----
    {
      const __bf16* pp = pb + (lane & 15) * 32 + ((lane < 16) ? 0 : 8);
      v8bf plo = *(const v8bf*)pp;
      v8bf phi = *(const v8bf*)(pp + 16);
      v16bf pf;
#pragma unroll
      for (int i = 0; i < 8; ++i) { pf[i] = plo[i]; pf[i + 8] = phi[i]; }

      accL = wmma_bf16(pf, onesf, accL);  // row sums: P @ ones
#pragma unroll
      for (int t = 0; t < 4; ++t)
        accO[t] = wmma_bf16(pf, load_frag(vc_base, vc_ld, t * 16, vc_k0), accO[t]);
    }
  }

  // ---- normalize and write context in (S,B,D) bf16 layout ----
#pragma unroll
  for (int t = 0; t < 4; ++t) {
    const int d = t * 16 + col;
#pragma unroll
    for (int r = 0; r < 8; ++r) {
      const int si = s0 + rbase + r;
      const float o = accO[t][r] / accL[r];
      Ctx[(size_t)(si * BATCH + b) * D_MODEL + h * D_K + d] = (__bf16)o;
    }
  }
}

// ---------------------------------------------------------------------------
// Host launch
// ---------------------------------------------------------------------------
extern "C" void kernel_launch(void* const* d_in, const int* in_sizes, int n_in,
                              void* d_out, int out_size, void* d_ws, size_t ws_size,
                              hipStream_t stream) {
  const float* q    = (const float*)d_in[0];
  const float* k    = (const float*)d_in[1];
  const float* v    = (const float*)d_in[2];
  const float* mask = (const float*)d_in[3];
  const float* Wq   = (const float*)d_in[4];
  const float* bq   = (const float*)d_in[5];
  const float* Wk   = (const float*)d_in[6];
  const float* bk   = (const float*)d_in[7];
  const float* Wv   = (const float*)d_in[8];
  const float* bv   = (const float*)d_in[9];
  const float* Wo   = (const float*)d_in[10];
  const float* bo   = (const float*)d_in[11];
  float* out        = (float*)d_out;

  const size_t NQ = (size_t)SEQ * BATCH * D_MODEL;  // 4M elements
  const size_t NW = (size_t)D_MODEL * D_MODEL;      // 1M elements

  __bf16* qb  = (__bf16*)d_ws;
  __bf16* kbw = qb + NQ;
  __bf16* vbw = kbw + NQ;
  __bf16* Wqb = vbw + NQ;
  __bf16* Wkb = Wqb + NW;
  __bf16* Wvb = Wkb + NW;
  __bf16* Wob = Wvb + NW;
  __bf16* Qh  = Wob + NW;
  __bf16* Kh  = Qh + NQ;
  __bf16* Vt  = Kh + NQ;
  __bf16* Ctx = Vt + NQ;  // total: 7*NQ + 4*NW bf16 = 64 MB

  f32_to_bf16<<<(int)(NQ / 256), 256, 0, stream>>>(q, qb, (int)NQ);
  f32_to_bf16<<<(int)(NQ / 256), 256, 0, stream>>>(k, kbw, (int)NQ);
  f32_to_bf16<<<(int)(NQ / 256), 256, 0, stream>>>(v, vbw, (int)NQ);
  f32_to_bf16<<<(int)(NW / 256), 256, 0, stream>>>(Wq, Wqb, (int)NW);
  f32_to_bf16<<<(int)(NW / 256), 256, 0, stream>>>(Wk, Wkb, (int)NW);
  f32_to_bf16<<<(int)(NW / 256), 256, 0, stream>>>(Wv, Wvb, (int)NW);
  f32_to_bf16<<<(int)(NW / 256), 256, 0, stream>>>(Wo, Wob, (int)NW);

  dim3 pgrid(SEQ / 256, D_MODEL / 64, BATCH);
  proj_kernel<0><<<pgrid, 256, 0, stream>>>(qb, Wqb, bq, (void*)Qh);
  proj_kernel<0><<<pgrid, 256, 0, stream>>>(kbw, Wkb, bk, (void*)Kh);
  proj_kernel<1><<<pgrid, 256, 0, stream>>>(vbw, Wvb, bv, (void*)Vt);

  dim3 agrid(SEQ / 64, BATCH * NHEAD);
  attn_kernel<<<agrid, 128, 0, stream>>>(Qh, Kh, Vt, mask, Ctx);

  dim3 ogrid((SEQ * BATCH) / 256, D_MODEL / 64, 1);
  proj_kernel<2><<<ogrid, 256, 0, stream>>>(Ctx, Wob, bo, (void*)out);
}
